// iRMB_87660282512040
// MI455X (gfx1250) — matrix operand order repaired and checked by reading, not compiled, not run on hardware
//
#include <hip/hip_runtime.h>
#include <hip/hip_bf16.h>

// ---- WMMA fragment types (wave32, gfx1250) ----
typedef __attribute__((ext_vector_type(16))) __bf16         v16bf;
typedef __attribute__((ext_vector_type(8)))  float          v8f;
typedef __attribute__((ext_vector_type(16))) unsigned short v16u;
typedef __attribute__((ext_vector_type(8)))  unsigned short v8u;

#define C_    128
#define HW    28
#define Ssp   21952          // 28^3
#define LW    343            // 7^3
#define LPAD  352            // 22 * 16
#define NTL   22
#define DH_   32
#define HEADS_ 4
#define SCALE_ 0.1767766952966369f   // 32^-0.5
#define EPS_  1e-6f

__device__ __forceinline__ unsigned short f2bfu(float x) {
  unsigned u = __builtin_bit_cast(unsigned, x);
  unsigned r = u + 0x7FFFu + ((u >> 16) & 1u);
  return (unsigned short)(r >> 16);
}

// A-matrix fragment (16x32 bf16): lane row = lane&15; K(e) = half*16 + e  (contiguous 32B)
__device__ __forceinline__ v16bf afrag(const unsigned short* rowp, int k0, int half) {
  v16u r = *(const v16u*)(rowp + k0 + half * 16);
  return __builtin_bit_cast(v16bf, r);
}

// B-matrix fragment (32x16 bf16): lane col = lane&15; K(e) = half*16 + e  (contiguous 32B)
__device__ __forceinline__ v16bf bfrag(const unsigned short* colp, int k0, int half) {
  v16u r = *(const v16u*)(colp + k0 + half * 16);
  return __builtin_bit_cast(v16bf, r);
}

__device__ __forceinline__ v8f wmma_bf16(v16bf a, v16bf b, v8f c) {
  return __builtin_amdgcn_wmma_f32_16x16x32_bf16(false, a, false, b, (short)0, c, false, false);
}

// low 32 bits of a generic pointer to LDS == byte offset within the workgroup LDS alloc
__device__ __forceinline__ unsigned lds_off(const void* p) {
  return (unsigned)(unsigned long long)p;
}

// gfx1250 async global->LDS copy, 16B per lane (CDNA5 ISA, GV mode)
#define ASYNC_LDS_B128(ldsoff, gaddr)                                                   \
  asm volatile("global_load_async_to_lds_b128 %0, %1, off" :: "v"(ldsoff), "v"(gaddr)   \
               : "memory")

// ---------------- weight prep: fp32 -> bf16 ----------------
// w_qk / w_v stay row-major [o][c]; w_cl and w_proj become K-major panels [ks][o][32]
// (one 8KB contiguous panel per 32-wide K-step -> clean async copy to LDS).
__global__ void prep_kernel(const float* __restrict__ wqk, const float* __restrict__ wv,
                            const float* __restrict__ wcl, const float* __restrict__ wproj,
                            unsigned short* wqk_bf, unsigned short* wv_bf,
                            unsigned short* wclB, unsigned short* wprojB) {
  int gid = blockIdx.x * 256 + threadIdx.x;
  if (gid < 256 * 128) wqk_bf[gid] = f2bfu(wqk[gid]);
  if (gid < 128 * 128) wv_bf[gid] = f2bfu(wv[gid]);
  if (gid < 128 * 3456) {                       // conv weights: k = tap*128 + i
    int ks = gid >> 12, rem = gid & 4095, o = rem >> 5, kk = rem & 31;
    int tap = ks >> 2, i = (ks & 3) * 32 + kk;
    wclB[gid] = f2bfu(wcl[(o * 128 + i) * 27 + tap]);
  }
  if (gid < 128 * 128) {                        // proj weights
    int ks = gid >> 12, rem = gid & 4095, o = rem >> 5, kk = rem & 31;
    wprojB[gid] = f2bfu(wproj[o * 128 + ks * 32 + kk]);
  }
}

// ---------------- per-channel mean / rstd over (B=4, C, 28^3) ----------------
__global__ void bn_stats_kernel(const float* __restrict__ x, float* mean, float* rstd) {
  int c = blockIdx.x;
  const float* p = x + (size_t)c * Ssp;
  float s = 0.f, s2 = 0.f;
  for (int n = threadIdx.x; n < 4 * Ssp; n += 256) {
    int bi = n / Ssp, sp = n % Ssp;
    float v = p[(size_t)bi * (C_ * Ssp) + sp];
    s += v; s2 += v * v;
  }
  __shared__ float sh[2][256];
  sh[0][threadIdx.x] = s; sh[1][threadIdx.x] = s2;
  __syncthreads();
  for (int st = 128; st > 0; st >>= 1) {
    if (threadIdx.x < st) {
      sh[0][threadIdx.x] += sh[0][threadIdx.x + st];
      sh[1][threadIdx.x] += sh[1][threadIdx.x + st];
    }
    __syncthreads();
  }
  if (threadIdx.x == 0) {
    float m = sh[0][0] * (1.0f / (4.0f * Ssp));
    float v = sh[1][0] * (1.0f / (4.0f * Ssp)) - m * m;
    mean[c] = m; rstd[c] = rsqrtf(v + EPS_);
  }
}

// ---------------- windowed attention, one block = (window, head) ----------------
__global__ void __launch_bounds__(256) attn_kernel(
    const float* __restrict__ x, const float* __restrict__ g_in, const float* __restrict__ b_in,
    const float* __restrict__ mean, const float* __restrict__ rstd,
    const unsigned short* __restrict__ wqk_bf, const unsigned short* __restrict__ wv_bf,
    float* __restrict__ xatt, unsigned short* __restrict__ xatt_bf) {
  extern __shared__ char smem[];
  unsigned short* xn  = (unsigned short*)smem;       // [352][128] normalized window (bf16)
  unsigned short* qL  = xn + LPAD * C_;              // [352][32]
  unsigned short* kL  = qL + LPAD * DH_;             // [352][32]
  unsigned short* vL  = kL + LPAD * DH_;             // [32][352]
  unsigned short* scr = vL + DH_ * LPAD;             // [8 waves][16*32] P transpose scratch

  const int tid = threadIdx.x;
  const int lane = tid & 31, wave = tid >> 5;
  const int half = lane >> 4, ln = lane & 15;

  const int blk = blockIdx.x;                // window id
  const int head = blockIdx.y;
  const int bi = blk >> 6, w1 = (blk >> 4) & 3, w2 = (blk >> 2) & 3, w3 = blk & 3;

  // zero l-padding rows 343..351
  for (int idx = tid; idx < (LPAD - LW) * C_; idx += 256) xn[LW * C_ + idx] = 0;

  // load + batchnorm the (dilated) 7x7x7 window
  for (int idx = tid; idx < LW * C_; idx += 256) {
    int l = idx >> 7, c = idx & 127;
    int i = l / 49, j = (l / 7) % 7, k = l % 7;
    int h = i * 4 + w1, w = j * 4 + w2, d = k * 4 + w3;
    float sc = g_in[c] * rstd[c];
    float sf = b_in[c] - mean[c] * sc;
    float v = x[((size_t)(bi * C_ + c) * HW + h) * (HW * HW) + (size_t)w * HW + d];
    xn[l * C_ + c] = f2bfu(v * sc + sf);
  }
  __syncthreads();

  // ---- QKV GEMM: 96 output rows (q|k|v per head) x 352 cols, K=128 ----
  // tile boundaries (16) align with q/k/v boundaries (32/64) -> branch once per tile
  for (int t = wave; t < 6 * NTL; t += 8) {
    int mt = t % 6, nt = t / 6;
    const unsigned short* wrow;
    if (mt < 2)      wrow = wqk_bf + (size_t)(head * 32 + mt * 16 + ln) * C_;
    else if (mt < 4) wrow = wqk_bf + (size_t)(128 + head * 32 + (mt - 2) * 16 + ln) * C_;
    else             wrow = wv_bf  + (size_t)(head * 32 + (mt - 4) * 16 + ln) * C_;
    v8f acc = {};
#pragma unroll
    for (int ks = 0; ks < 4; ++ks) {
      v16bf a = afrag(wrow, ks * 32, half);
      v16bf b = bfrag(xn + (nt * 16 + ln) * C_, ks * 32, half);
      acc = wmma_bf16(a, b, acc);
    }
    int l = nt * 16 + ln;
    int dd0 = (mt & 1) * 16 + 8 * half;          // row-within-(q|k|v)
    if (mt < 4) {
      unsigned short* dst = (mt < 2) ? qL : kL;
#pragma unroll
      for (int r = 0; r < 8; ++r) dst[l * DH_ + dd0 + r] = f2bfu(acc[r]);
    } else {
#pragma unroll
      for (int r = 0; r < 8; ++r)
        vL[(dd0 + r) * LPAD + l] = f2bfu(fmaxf(acc[r], 0.0f));   // ReLU(v)
    }
  }
  __syncthreads();

  // ---- flash attention over l-tiles (per wave) ----
  unsigned short* myscr = scr + wave * (16 * 32);
  for (int lt = wave; lt < NTL; lt += 8) {
    v8f o0 = {}, o1 = {};
    float rmax[8], rsum[8];
#pragma unroll
    for (int r = 0; r < 8; ++r) { rmax[r] = -3.0e38f; rsum[r] = 0.0f; }
    v16bf qa = afrag(qL + (lt * 16 + ln) * DH_, 0, half);

    for (int mc = 0; mc < 11; ++mc) {
      int m0 = mc * 32;
      v16bf kb0 = bfrag(kL + (m0 + ln) * DH_, 0, half);
      v16bf kb1 = bfrag(kL + (m0 + 16 + ln) * DH_, 0, half);
      v8f zc = {};
      v8f s0 = wmma_bf16(qa, kb0, zc);
      v8f s1 = wmma_bf16(qa, kb1, zc);
      bool ok0 = (m0 + ln) < LW, ok1 = (m0 + 16 + ln) < LW;
#pragma unroll
      for (int r = 0; r < 8; ++r) {
        float a0 = ok0 ? s0[r] * SCALE_ : -3.0e38f;
        float a1 = ok1 ? s1[r] * SCALE_ : -3.0e38f;
        float mx = fmaxf(a0, a1);
        mx = fmaxf(mx, __shfl_xor(mx, 1)); mx = fmaxf(mx, __shfl_xor(mx, 2));
        mx = fmaxf(mx, __shfl_xor(mx, 4)); mx = fmaxf(mx, __shfl_xor(mx, 8));
        float nm = fmaxf(rmax[r], mx);
        float alpha = __expf(rmax[r] - nm);
        float p0 = ok0 ? __expf(a0 - nm) : 0.0f;
        float p1 = ok1 ? __expf(a1 - nm) : 0.0f;
        float ps = p0 + p1;
        ps += __shfl_xor(ps, 1); ps += __shfl_xor(ps, 2);
        ps += __shfl_xor(ps, 4); ps += __shfl_xor(ps, 8);
        rsum[r] = rsum[r] * alpha + ps;
        rmax[r] = nm;
        o0[r] *= alpha; o1[r] *= alpha;
        myscr[(r + 8 * half) * 32 + ln]      = f2bfu(p0);
        myscr[(r + 8 * half) * 32 + 16 + ln] = f2bfu(p1);
      }
      v16bf pa  = afrag(myscr + ln * 32, 0, half);
      v16bf vb0 = bfrag(vL + ln * LPAD, m0, half);
      v16bf vb1 = bfrag(vL + (16 + ln) * LPAD, m0, half);
      o0 = wmma_bf16(pa, vb0, o0);
      o1 = wmma_bf16(pa, vb1, o1);
    }

    // normalize + scatter back to un-windowed (B,C,H,W,D)
    int lrow_base = lt * 16 + 8 * half;
#pragma unroll
    for (int r = 0; r < 8; ++r) {
      int l = lrow_base + r;
      if (l >= LW) continue;
      float inv = 1.0f / rsum[r];
      int i = l / 49, j = (l / 7) % 7, k = l % 7;
      int h = i * 4 + w1, w = j * 4 + w2, d = k * 4 + w3;
      size_t spat = ((size_t)h * HW + w) * HW + d;
      size_t a0 = (size_t)(bi * C_ + head * DH_ + ln) * Ssp + spat;
      size_t a1 = (size_t)(bi * C_ + head * DH_ + 16 + ln) * Ssp + spat;
      float v0 = o0[r] * inv, v1 = o1[r] * inv;
      xatt[a0] = v0;            xatt[a1] = v1;
      xatt_bf[a0] = f2bfu(v0);  xatt_bf[a1] = f2bfu(v1);
    }
  }
}

// ---------------- 3x3x3 conv as implicit GEMM: M=128, N=28 (d line), K=27*128 ----------------
// A-panels (8KB/K-step) double-buffered into LDS with global_load_async_to_lds_b128.
__global__ void __launch_bounds__(256) conv3_kernel(
    const unsigned short* __restrict__ xbf, const unsigned short* __restrict__ wclB,
    float* __restrict__ y) {
  __shared__ unsigned short Ap[2][128 * 32];        // weight panels  [o][kk]
  __shared__ unsigned short Bp[32 * 32];            // [d 0..31][i 0..31] bf16
  const int tid = threadIdx.x, lane = tid & 31, wave = tid >> 5;
  const int half = lane >> 4, ln = lane & 15;
  const int h = blockIdx.x / HW, w = blockIdx.x % HW, b = blockIdx.y;

  if (tid < 128) Bp[(28 + (tid >> 5)) * 32 + (tid & 31)] = 0;   // d padding rows

  // precompute this thread's B-staging coordinates (32ch x 28d = 896 elements)
  int il_[4], dp_[4], nst = 0;
  for (int e = tid; e < 32 * 28; e += 256) { il_[nst] = e / 28; dp_[nst] = e % 28; ++nst; }

  v8f acc0 = {}, acc1 = {};

  // issue async copy of weight panel `ks` into buffer `buf` (2 x b128 per lane)
  auto issue_panel = [&](int ks, int buf) {
    unsigned l0 = lds_off(&Ap[buf][0]);
    const unsigned short* gp = wclB + (size_t)ks * 4096;
#pragma unroll
    for (int j = 0; j < 2; ++j) {
      unsigned off = (unsigned)(tid + j * 256) * 8;      // ushort units, 16B chunks
      ASYNC_LDS_B128(l0 + off * 2, gp + off);
    }
  };

  issue_panel(0, 0);

  for (int ks = 0; ks < 108; ++ks) {
    int buf = ks & 1;
    int tap = ks >> 2, i0 = (ks & 3) * 32;
    int kh = tap / 9, kw = (tap / 3) % 3, kd = tap % 3;
    int hh = h + kh - 1, ww = w + kw - 1;
    bool inplane = (hh >= 0) && (hh < HW) && (ww >= 0) && (ww < HW);
    __syncthreads();                                   // Bp / Ap[buf^1] consumers done
    for (int j = 0; j < nst; ++j) {
      int dd = dp_[j] + kd - 1;
      unsigned short v = 0;
      if (inplane && dd >= 0 && dd < HW)
        v = xbf[((size_t)(b * C_ + i0 + il_[j]) * HW + hh) * (HW * HW) + (size_t)ww * HW + dd];
      Bp[dp_[j] * 32 + il_[j]] = v;
    }
    if (ks + 1 < 108) {
      issue_panel(ks + 1, buf ^ 1);                    // overlap next panel copy
      asm volatile("s_wait_asynccnt 0x2" ::: "memory"); // current panel landed
    } else {
      asm volatile("s_wait_asynccnt 0x0" ::: "memory");
    }
    __syncthreads();
    v16bf a  = afrag(&Ap[buf][(wave * 16 + ln) * 32], 0, half);
    v16bf b0 = bfrag(Bp + ln * 32, 0, half);
    v16bf b1 = bfrag(Bp + (16 + ln) * 32, 0, half);
    acc0 = wmma_bf16(a, b0, acc0);
    acc1 = wmma_bf16(a, b1, acc1);
  }
#pragma unroll
  for (int r = 0; r < 8; ++r) {
    int o = wave * 16 + r + 8 * half;
    size_t base = ((size_t)(b * C_ + o) * HW + h) * (HW * HW) + (size_t)w * HW;
    y[base + ln] = acc0[r];
    if (16 + ln < HW) y[base + 16 + ln] = acc1[r];
  }
}

// ---------------- epilogue: t = xatt + silu(bn(y)); out = shortcut + w_proj @ t ----------------
__global__ void __launch_bounds__(256) epi_kernel(
    const float* __restrict__ xatt, const float* __restrict__ y,
    const float* __restrict__ g_cl, const float* __restrict__ b_cl,
    const float* __restrict__ mean_y, const float* __restrict__ rstd_y,
    const unsigned short* __restrict__ wprojB,
    const float* __restrict__ shortcut, float* __restrict__ out) {
  __shared__ unsigned short ApE[128 * 32];
  __shared__ unsigned short Tp[32 * 32];
  const int tid = threadIdx.x, lane = tid & 31, wave = tid >> 5;
  const int half = lane >> 4, ln = lane & 15;
  const int h = blockIdx.x / HW, w = blockIdx.x % HW, b = blockIdx.y;
  if (tid < 128) Tp[(28 + (tid >> 5)) * 32 + (tid & 31)] = 0;

  v8f acc0 = {}, acc1 = {};

  for (int ks = 0; ks < 4; ++ks) {
    int i0 = ks * 32;
    __syncthreads();
    {                                                  // async-copy proj weight panel
      unsigned l0 = lds_off(&ApE[0]);
      const unsigned short* gp = wprojB + (size_t)ks * 4096;
#pragma unroll
      for (int j = 0; j < 2; ++j) {
        unsigned off = (unsigned)(tid + j * 256) * 8;
        ASYNC_LDS_B128(l0 + off * 2, gp + off);
      }
    }
    for (int e = tid; e < 32 * 28; e += 256) {
      int il = e / 28, d = e % 28;
      int c = i0 + il;
      size_t ga = ((size_t)(b * C_ + c) * HW + h) * (HW * HW) + (size_t)w * HW + d;
      float sc = g_cl[c] * rstd_y[c];
      float sf = b_cl[c] - mean_y[c] * sc;
      float z = y[ga] * sc + sf;
      float t = xatt[ga] + z / (1.0f + __expf(-z));     // x + silu(bn(y))
      Tp[d * 32 + il] = f2bfu(t);
    }
    asm volatile("s_wait_asynccnt 0x0" ::: "memory");
    __syncthreads();
    v16bf a  = afrag(ApE + (wave * 16 + ln) * 32, 0, half);
    v16bf b0 = bfrag(Tp + ln * 32, 0, half);
    v16bf b1 = bfrag(Tp + (16 + ln) * 32, 0, half);
    acc0 = wmma_bf16(a, b0, acc0);
    acc1 = wmma_bf16(a, b1, acc1);
  }
#pragma unroll
  for (int r = 0; r < 8; ++r) {
    int o = wave * 16 + r + 8 * half;
    size_t base = ((size_t)(b * C_ + o) * HW + h) * (HW * HW) + (size_t)w * HW;
    out[base + ln] = shortcut[base + ln] + acc0[r];
    if (16 + ln < HW) out[base + 16 + ln] = shortcut[base + 16 + ln] + acc1[r];
  }
}

extern "C" void kernel_launch(void* const* d_in, const int* in_sizes, int n_in,
                              void* d_out, int out_size, void* d_ws, size_t ws_size,
                              hipStream_t stream) {
  (void)in_sizes; (void)n_in; (void)out_size; (void)ws_size;
  const float* x      = (const float*)d_in[0];
  const float* g_in   = (const float*)d_in[1];
  const float* b_in   = (const float*)d_in[2];
  const float* w_qk   = (const float*)d_in[3];
  const float* w_v    = (const float*)d_in[4];
  const float* w_cl   = (const float*)d_in[5];
  const float* g_cl   = (const float*)d_in[6];
  const float* b_cl   = (const float*)d_in[7];
  const float* w_proj = (const float*)d_in[8];
  float* out = (float*)d_out;
  char* ws = (char*)d_ws;

  // workspace layout (~108.2 MB total)
  float* mean_in = (float*)(ws + 0);
  float* rstd_in = mean_in + 128;
  float* mean_y  = rstd_in + 128;
  float* rstd_y  = mean_y + 128;
  unsigned short* wqk_bf = (unsigned short*)(ws + 2048);      // 64 KB
  unsigned short* wv_bf  = (unsigned short*)(ws + 67584);     // 32 KB
  unsigned short* wprojB = (unsigned short*)(ws + 100352);    // 32 KB (K-major panels)
  unsigned short* wclB   = (unsigned short*)(ws + 133120);    // 864 KB (K-major panels)
  float*          xatt    = (float*)(ws + (size_t)(1u << 20));           // 42.9 MB fp32
  unsigned short* xatt_bf = (unsigned short*)(ws + (size_t)46006272u);   // 21.4 MB bf16
  float*          ybuf    = (float*)(ws + (size_t)68485120u);            // 42.9 MB fp32

  prep_kernel<<<1728, 256, 0, stream>>>(w_qk, w_v, w_cl, w_proj, wqk_bf, wv_bf, wclB, wprojB);
  bn_stats_kernel<<<128, 256, 0, stream>>>(x, mean_in, rstd_in);

  size_t attn_lds = (size_t)(LPAD * C_ + 3 * LPAD * DH_ + 8 * 16 * 32) * 2;  // 165,888 B
  attn_kernel<<<dim3(256, HEADS_), 256, attn_lds, stream>>>(
      x, g_in, b_in, mean_in, rstd_in, wqk_bf, wv_bf, xatt, xatt_bf);

  conv3_kernel<<<dim3(HW * HW, 4), 256, 0, stream>>>(xatt_bf, wclB, ybuf);
  bn_stats_kernel<<<128, 256, 0, stream>>>(ybuf, mean_y, rstd_y);
  epi_kernel<<<dim3(HW * HW, 4), 256, 0, stream>>>(
      xatt, ybuf, g_cl, b_cl, mean_y, rstd_y, wprojB, x, out);
}
